// ALBertQALayer_49185965474528
// MI455X (gfx1250) — matrix operand order repaired and checked by reading, not compiled, not run on hardware
//
#include <hip/hip_runtime.h>
#include <hip/hip_bf16.h>
#include <math.h>

// ---------------------------------------------------------------------------
// Types for CDNA5 WMMA (wave32): v16bf A/B fragments, v8f accumulator
// ---------------------------------------------------------------------------
typedef __attribute__((ext_vector_type(16))) __bf16 v16bf;
typedef __attribute__((ext_vector_type(8)))  float  v8f;

union FragU {
    uint4 q[2];
    unsigned short us[16];
    v16bf v;
};

static __device__ __forceinline__ unsigned short f2bf(float f) {
    unsigned int u = __float_as_uint(f);
    unsigned int lsb = (u >> 16) & 1u;
    u += 0x7fffu + lsb;               // round-to-nearest-even
    return (unsigned short)(u >> 16);
}

static __device__ __forceinline__ float wave_sum(float v) {
    #pragma unroll
    for (int o = 16; o > 0; o >>= 1) v += __shfl_xor(v, o, 32);
    return v;
}

// CDNA5 async global->LDS copy (ASYNCcnt path), per CDNA5 ISA.
// vdst = VGPR with LDS byte address, vaddr = 64-bit global address, no SADDR.
static __device__ __forceinline__ void async_ld_b128(unsigned lds_off, const void* gptr) {
    asm volatile("global_load_async_to_lds_b128 %0, %1, off"
                 :: "v"(lds_off), "v"(gptr) : "memory");
}
static __device__ __forceinline__ void wait_async0() {
#if __has_builtin(__builtin_amdgcn_s_wait_asynccnt)
    __builtin_amdgcn_s_wait_asynccnt(0);
#else
    asm volatile("s_wait_asynccnt 0x0" ::: "memory");
#endif
}

#define Bb 16
#define Ss 512
#define Hh 1024
#define NEGINF (-1e30f)

// ---------------------------------------------------------------------------
// K0a: seq fp32 -> bf16 (row-major, GEMM A operand)
// ---------------------------------------------------------------------------
__global__ __launch_bounds__(256) void k_xbf(
    const float* __restrict__ X, unsigned short* __restrict__ Xb)
{
    size_t i = ((size_t)blockIdx.x * 256 + threadIdx.x) * 4;
    float4 f = *(const float4*)(X + i);
    uint2 o;
    o.x = (unsigned)f2bf(f.x) | ((unsigned)f2bf(f.y) << 16);
    o.y = (unsigned)f2bf(f.z) | ((unsigned)f2bf(f.w) << 16);
    *(uint2*)(Xb + i) = o;
}

// ---------------------------------------------------------------------------
// K0b: W0a fp32 (d-major) -> bf16 transposed Wt[h][d]  (GEMM B operand,
// k-contiguous so tiles are raw byte copies -> async-to-LDS friendly)
// ---------------------------------------------------------------------------
__global__ __launch_bounds__(256) void k_wt(
    const float* __restrict__ W, unsigned short* __restrict__ Wt)
{
    __shared__ unsigned short t[32][33];
    int h0 = blockIdx.x * 32, d0 = blockIdx.y * 32;
    int r = threadIdx.x >> 5, c = threadIdx.x & 31;
    #pragma unroll
    for (int i = 0; i < 4; ++i) {
        int dd = r + i * 8;
        t[dd][c] = f2bf(W[(size_t)(d0 + dd) * Hh + h0 + c]);
    }
    __syncthreads();
    #pragma unroll
    for (int i = 0; i < 4; ++i) {
        int hh = r + i * 8;
        Wt[(size_t)(h0 + hh) * Hh + d0 + c] = t[c][hh];
    }
}

// ---------------------------------------------------------------------------
// K1: start_logits[b,s] = seq[b,s,:] . w_start + b_start   (wave per row)
// ---------------------------------------------------------------------------
__global__ __launch_bounds__(256) void k_start_logits(
    const float* __restrict__ X, const float* __restrict__ w,
    const float* __restrict__ b, float* __restrict__ logits)
{
    int row  = blockIdx.x * 8 + (threadIdx.x >> 5);
    int lane = threadIdx.x & 31;
    const float* x = X + (size_t)row * Hh;
    float acc = 0.f;
    #pragma unroll 8
    for (int j = lane; j < Hh; j += 32) acc += x[j] * w[j];
    acc = wave_sum(acc);
    if (lane == 0) logits[row] = acc + b[0];
}

// ---------------------------------------------------------------------------
// K2: per-batch mask + log_softmax + top5 + start_feature + gather top rows
// ---------------------------------------------------------------------------
__global__ __launch_bounds__(512) void k_start_softmax(
    const float* __restrict__ X, const float* __restrict__ pmask,
    const float* __restrict__ logits, float* __restrict__ out,
    float* __restrict__ sfeat, float* __restrict__ sgath)
{
    __shared__ float sm[Ss];
    __shared__ float sp[Ss];
    __shared__ float red[Ss];
    __shared__ int   redi[Ss];
    __shared__ float s_lse;
    __shared__ int   topidx[5];

    int b = blockIdx.x, s = threadIdx.x;
    float pm = pmask[b * Ss + s];
    float m  = logits[b * Ss + s] * (1.f - pm) + NEGINF * pm;
    sm[s] = m;
    red[s] = m; __syncthreads();
    for (int off = 256; off > 0; off >>= 1) {
        if (s < off) red[s] = fmaxf(red[s], red[s + off]);
        __syncthreads();
    }
    float mx = red[0]; __syncthreads();
    red[s] = __expf(m - mx); __syncthreads();
    for (int off = 256; off > 0; off >>= 1) {
        if (s < off) red[s] += red[s + off];
        __syncthreads();
    }
    if (s == 0) s_lse = mx + __logf(red[0]);
    __syncthreads();
    float lse = s_lse;
    sp[s] = __expf(m - lse);

    for (int t = 0; t < 5; ++t) {
        red[s] = sm[s]; redi[s] = s; __syncthreads();
        for (int off = 256; off > 0; off >>= 1) {
            if (s < off) {
                if (red[s+off] > red[s] || (red[s+off] == red[s] && redi[s+off] < redi[s])) {
                    red[s] = red[s+off]; redi[s] = redi[s+off];
                }
            }
            __syncthreads();
        }
        if (s == 0) {
            topidx[t] = redi[0];
            out[b * 5 + t]      = red[0] - lse;      // start_top_log_probs
            out[80 + b * 5 + t] = (float)redi[0];    // start_top_index
        }
        __syncthreads();
        if (s == topidx[t]) sm[s] = -INFINITY;
        __syncthreads();
    }

    const float* xb = X + (size_t)b * Ss * Hh;
    for (int h = s; h < Hh; h += Ss) {
        float acc = 0.f;
        for (int ss = 0; ss < Ss; ++ss) acc += sp[ss] * xb[(size_t)ss * Hh + h];
        sfeat[b * Hh + h] = acc;
        #pragma unroll
        for (int k = 0; k < 5; ++k)
            sgath[((size_t)b * 5 + k) * Hh + h] = xb[(size_t)topidx[k] * Hh + h];
    }
}

// ---------------------------------------------------------------------------
// K3: v[b,k,h] = b_end0[h] + sum_d sgath[b,k,d] * w_end0[H+d, h]
// ---------------------------------------------------------------------------
__global__ __launch_bounds__(256) void k_vrows(
    const float* __restrict__ sgath, const float* __restrict__ W,
    const float* __restrict__ b0, float* __restrict__ V)
{
    int bk = blockIdx.x;
    __shared__ float sf[Hh];
    for (int d = threadIdx.x; d < Hh; d += 256) sf[d] = sgath[(size_t)bk * Hh + d];
    __syncthreads();
    for (int h = threadIdx.x; h < Hh; h += 256) {
        float acc = b0[h];
        const float* wp = W + (size_t)Hh * Hh + h;   // W0b column h
        for (int d = 0; d < Hh; ++d) acc += sf[d] * wp[(size_t)d * Hh];
        V[(size_t)bk * Hh + h] = acc;
    }
}

// ---------------------------------------------------------------------------
// K4: U = seq @ W0a via v_wmma_f32_16x16x32_bf16.
// 64x64 tile / 256 threads, 8 waves = 4(M) x 2(N); K-step 64 => 4 WMMAs per
// barrier pair. Tiles staged with global_load_async_to_lds_b128 (ASYNCcnt),
// fragments built K-contiguous (k = 16*half + s) with ds_load_b128.
// ---------------------------------------------------------------------------
__global__ __launch_bounds__(256) void k_gemm_u(
    const unsigned short* __restrict__ Xbf, const unsigned short* __restrict__ Wt,
    float* __restrict__ U)
{
    __shared__ __align__(16) unsigned short sA[64][64];   // [m][k] bf16
    __shared__ __align__(16) unsigned short sB[64][64];   // [n][k] bf16

    const int tid  = threadIdx.x;
    const int wave = tid >> 5, lane = tid & 31;
    const int mi = wave & 3, ni = wave >> 2;
    const int bm = blockIdx.x * 64;
    const int bn = blockIdx.y * 64;

    v8f acc0 = {}; v8f acc1 = {};

    // copy mapping: thread t moves two 16B segments of row (t>>2), cols seg..seg+15
    const int lrow = tid >> 2;
    const int lseg = (tid & 3) * 16;
    const unsigned aoff0 = (unsigned)(size_t)&sA[lrow][lseg];
    const unsigned aoff1 = (unsigned)(size_t)&sA[lrow][lseg + 8];
    const unsigned boff0 = (unsigned)(size_t)&sB[lrow][lseg];
    const unsigned boff1 = (unsigned)(size_t)&sB[lrow][lseg + 8];
    const unsigned short* ag = Xbf + (size_t)(bm + lrow) * Hh + lseg;
    const unsigned short* bg = Wt  + (size_t)(bn + lrow) * Hh + lseg;

    const int half = lane >> 4;
    const int am   = mi * 16 + (lane & 15);
    const int bn0  = ni * 32 + (lane & 15);

    for (int kt = 0; kt < Hh; kt += 64) {
        async_ld_b128(aoff0, ag + kt);
        async_ld_b128(aoff1, ag + kt + 8);
        async_ld_b128(boff0, bg + kt);
        async_ld_b128(boff1, bg + kt + 8);
        wait_async0();
        __syncthreads();

        #pragma unroll
        for (int kk = 0; kk < 64; kk += 32) {
            FragU a, b0f, b1f;
            // A 16x32 bf16: lanes0-15 K=0..15, lanes16-31 K=16..31 (K contiguous)
            a.q[0] = *(const uint4*)&sA[am][kk + half * 16];
            a.q[1] = *(const uint4*)&sA[am][kk + half * 16 + 8];
            // B 32x16 bf16: lanes0-15 K=0..15, lanes16-31 K=16..31 (K contiguous)
            b0f.q[0] = *(const uint4*)&sB[bn0][kk + half * 16];
            b0f.q[1] = *(const uint4*)&sB[bn0][kk + half * 16 + 8];
            b1f.q[0] = *(const uint4*)&sB[bn0 + 16][kk + half * 16];
            b1f.q[1] = *(const uint4*)&sB[bn0 + 16][kk + half * 16 + 8];

            acc0 = __builtin_amdgcn_wmma_f32_16x16x32_bf16(
                false, a.v, false, b0f.v, (short)0, acc0, false, false);
            acc1 = __builtin_amdgcn_wmma_f32_16x16x32_bf16(
                false, a.v, false, b1f.v, (short)0, acc1, false, false);
        }
        __syncthreads();
    }

    // C/D layout: VGPR r -> M = r + 8*(lane/16), N = lane%16
    const int col0 = bn + ni * 32 + (lane & 15);
    #pragma unroll
    for (int r = 0; r < 8; ++r) {
        int row = bm + mi * 16 + r + half * 8;
        U[(size_t)row * Hh + col0]      = acc0[r];
        U[(size_t)row * Hh + col0 + 16] = acc1[r];
    }
}

// ---------------------------------------------------------------------------
// K5: fused tanh + LayerNorm + dot(w_end1): EL[b,k,s]  (block per (b,s))
// ---------------------------------------------------------------------------
__global__ __launch_bounds__(256) void k_end_logits(
    const float* __restrict__ U, const float* __restrict__ V,
    const float* __restrict__ g, const float* __restrict__ bl,
    const float* __restrict__ w1, const float* __restrict__ b1,
    float* __restrict__ EL)
{
    int bs = blockIdx.x;
    int b = bs >> 9, s = bs & 511;
    __shared__ float su[Hh];
    __shared__ float red[256];
    int tid = threadIdx.x;
    for (int h = tid; h < Hh; h += 256) su[h] = U[(size_t)bs * Hh + h];
    __syncthreads();

    for (int k = 0; k < 5; ++k) {
        const float* vk = V + ((size_t)b * 5 + k) * Hh;
        float t[4], sum = 0.f, sumsq = 0.f;
        #pragma unroll
        for (int i = 0; i < 4; ++i) {
            int h = tid + i * 256;
            float e = tanhf(su[h] + vk[h]);
            t[i] = e; sum += e; sumsq += e * e;
        }
        red[tid] = sum; __syncthreads();
        for (int off = 128; off > 0; off >>= 1) { if (tid < off) red[tid] += red[tid + off]; __syncthreads(); }
        float mu = red[0] * (1.f / Hh); __syncthreads();
        red[tid] = sumsq; __syncthreads();
        for (int off = 128; off > 0; off >>= 1) { if (tid < off) red[tid] += red[tid + off]; __syncthreads(); }
        float var = red[0] * (1.f / Hh) - mu * mu; __syncthreads();
        float rs = rsqrtf(var + 1e-12f);
        float part = 0.f;
        #pragma unroll
        for (int i = 0; i < 4; ++i) {
            int h = tid + i * 256;
            part += ((t[i] - mu) * rs * g[h] + bl[h]) * w1[h];
        }
        red[tid] = part; __syncthreads();
        for (int off = 128; off > 0; off >>= 1) { if (tid < off) red[tid] += red[tid + off]; __syncthreads(); }
        if (tid == 0) EL[((size_t)b * 5 + k) * Ss + s] = red[0] + b1[0];
        __syncthreads();
    }
}

// ---------------------------------------------------------------------------
// K6: per-(b,k) masked log_softmax over S + top5
// ---------------------------------------------------------------------------
__global__ __launch_bounds__(512) void k_end_softmax(
    const float* __restrict__ EL, const float* __restrict__ pmask,
    float* __restrict__ out)
{
    __shared__ float sm[Ss];
    __shared__ float red[Ss];
    __shared__ int   redi[Ss];
    __shared__ float s_lse;
    __shared__ int   chosen;
    int bk = blockIdx.x, b = bk / 5, s = threadIdx.x;
    float pm = pmask[b * Ss + s];
    float m  = EL[(size_t)bk * Ss + s] * (1.f - pm) + NEGINF * pm;
    sm[s] = m;
    red[s] = m; __syncthreads();
    for (int off = 256; off > 0; off >>= 1) { if (s < off) red[s] = fmaxf(red[s], red[s + off]); __syncthreads(); }
    float mx = red[0]; __syncthreads();
    red[s] = __expf(m - mx); __syncthreads();
    for (int off = 256; off > 0; off >>= 1) { if (s < off) red[s] += red[s + off]; __syncthreads(); }
    if (s == 0) s_lse = mx + __logf(red[0]);
    __syncthreads();
    float lse = s_lse;
    for (int t = 0; t < 5; ++t) {
        red[s] = sm[s]; redi[s] = s; __syncthreads();
        for (int off = 256; off > 0; off >>= 1) {
            if (s < off) {
                if (red[s+off] > red[s] || (red[s+off] == red[s] && redi[s+off] < redi[s])) {
                    red[s] = red[s+off]; redi[s] = redi[s+off];
                }
            }
            __syncthreads();
        }
        if (s == 0) {
            chosen = redi[0];
            out[160 + bk * 5 + t] = red[0] - lse;     // end_top_log_probs
            out[560 + bk * 5 + t] = (float)redi[0];   // end_top_index
        }
        __syncthreads();
        if (s == chosen) sm[s] = -INFINITY;
        __syncthreads();
    }
}

// ---------------------------------------------------------------------------
// K7: cls head: tanh([start_feature, cls_feature] @ w_ans0 + b_ans0) @ w_ans1
// ---------------------------------------------------------------------------
__global__ __launch_bounds__(256) void k_cls(
    const float* __restrict__ X, const float* __restrict__ sfeat,
    const float* __restrict__ w0, const float* __restrict__ b0,
    const float* __restrict__ w1, float* __restrict__ out)
{
    int b = blockIdx.x, tid = threadIdx.x;
    __shared__ float f[2 * Hh];
    __shared__ float red[256];
    for (int d = tid; d < Hh; d += 256) {
        f[d]      = sfeat[b * Hh + d];
        f[Hh + d] = X[(size_t)b * Ss * Hh + d];   // seq[b,0,:]
    }
    __syncthreads();
    float part = 0.f;
    for (int h = tid; h < Hh; h += 256) {
        float acc = b0[h];
        for (int d = 0; d < 2 * Hh; ++d) acc += f[d] * w0[(size_t)d * Hh + h];
        part += tanhf(acc) * w1[h];
    }
    red[tid] = part; __syncthreads();
    for (int off = 128; off > 0; off >>= 1) { if (tid < off) red[tid] += red[tid + off]; __syncthreads(); }
    if (tid == 0) out[960 + b] = red[0];
}

// ---------------------------------------------------------------------------
extern "C" void kernel_launch(void* const* d_in, const int* in_sizes, int n_in,
                              void* d_out, int out_size, void* d_ws, size_t ws_size,
                              hipStream_t stream) {
    const float* seq     = (const float*)d_in[0];
    const float* p_mask  = (const float*)d_in[1];
    const float* w_start = (const float*)d_in[2];
    const float* b_start = (const float*)d_in[3];
    const float* w_end0  = (const float*)d_in[4];
    const float* b_end0  = (const float*)d_in[5];
    const float* ln_g    = (const float*)d_in[6];
    const float* ln_b    = (const float*)d_in[7];
    const float* w_end1  = (const float*)d_in[8];
    const float* b_end1  = (const float*)d_in[9];
    const float* w_ans0  = (const float*)d_in[10];
    const float* b_ans0  = (const float*)d_in[11];
    const float* w_ans1  = (const float*)d_in[12];
    (void)in_sizes; (void)n_in; (void)out_size; (void)ws_size;

    float* out = (float*)d_out;
    float* ws  = (float*)d_ws;

    float* logits = ws;                        // 8192 f32
    float* sfeat  = logits + Bb * Ss;          // 16384 f32
    float* sgath  = sfeat + Bb * Hh;           // 81920 f32
    float* V      = sgath + Bb * 5 * Hh;       // 81920 f32
    float* U      = V + Bb * 5 * Hh;           // 8388608 f32
    float* EL     = U + (size_t)Bb * Ss * Hh;  // 40960 f32
    unsigned short* Xbf = (unsigned short*)(EL + Bb * 5 * Ss);   // 8M bf16
    unsigned short* Wt  = Xbf + (size_t)Bb * Ss * Hh;            // 1M bf16

    k_xbf<<<dim3((Bb * Ss * Hh) / 1024), dim3(256), 0, stream>>>(seq, Xbf);
    k_wt<<<dim3(Hh / 32, Hh / 32), dim3(256), 0, stream>>>(w_end0, Wt);
    k_start_logits<<<dim3(Bb * Ss / 8), dim3(256), 0, stream>>>(seq, w_start, b_start, logits);
    k_start_softmax<<<dim3(Bb), dim3(Ss), 0, stream>>>(seq, p_mask, logits, out, sfeat, sgath);
    k_vrows<<<dim3(Bb * 5), dim3(256), 0, stream>>>(sgath, w_end0, b_end0, V);
    k_gemm_u<<<dim3(Bb * Ss / 64, Hh / 64), dim3(256), 0, stream>>>(Xbf, Wt, U);
    k_end_logits<<<dim3(Bb * Ss), dim3(256), 0, stream>>>(U, V, ln_g, ln_b, w_end1, b_end1, EL);
    k_end_softmax<<<dim3(Bb * 5), dim3(Ss), 0, stream>>>(EL, p_mask, out);
    k_cls<<<dim3(Bb), dim3(256), 0, stream>>>(seq, sfeat, w_ans0, b_ans0, w_ans1, out);
}
